// rnn_test_25494925869756
// MI455X (gfx1250) — compile-verified, matrix operand order checked by reading, not run on hardware
//
#include <hip/hip_runtime.h>

// ---------------- problem constants (match reference) ----------------
#define BATCH    256
#define TSTEPS   1024
#define H1N      64
#define H2N      256
#define OUTN     10
#define KTOT     (H1N + H2N)    // 320 : concat [s1, s2] feeding layer 2
#define TILE_B   16             // WMMA M tile; one workgroup per 16 batch rows
#define NWAVES   4
#define NTHREADS (NWAVES * 32)

typedef _Float16 v8h  __attribute__((ext_vector_type(8)));
typedef _Float16 v16h __attribute__((ext_vector_type(16)));
typedef float    v4f  __attribute__((ext_vector_type(4)));
typedef float    v8f  __attribute__((ext_vector_type(8)));

// Load one 16x32 f16 fragment (A or B operand) from an LDS tile stored
// row-major [row][k] ('stride' halves per row).
// ISA 7.12.2 (16-bit A 16x32): lanes 0-15 hold row=lane with K 0..7 in
// v0..3 and K 16..23 in v4..7; lanes 16-31 hold the same rows with K 8..15
// and K 24..31. B fragments stored [n][k] (i.e. W row-major, since
// B[k][n] = W[n][k]) use identical per-lane addressing -> two ds_load_b128.
__device__ __forceinline__ v16h ld_frag(const _Float16* base,
                                        int stride, int rowBase, int kBase,
                                        int lane) {
    const int r  = rowBase + (lane & 15);
    const int ko = kBase + ((lane & 16) ? 8 : 0);
    const _Float16* p = base + r * stride + ko;
    v8h lo = *(const v8h*)(p);
    v8h hi = *(const v8h*)(p + 16);
    return __builtin_shufflevector(lo, hi,
        0,1,2,3,4,5,6,7,8,9,10,11,12,13,14,15);
}

__device__ __forceinline__ v8f wmma16(v16h a, v16h b, v8f c) {
    // 8 args: (neg_a, A, neg_b, B, c_mod, C, reuse_a, reuse_b)
    return __builtin_amdgcn_wmma_f32_16x16x32_f16(false, a, false, b,
                                                  (short)0, c, false, false);
}

__global__ void __launch_bounds__(NTHREADS)
snn_forward_kernel(const float* __restrict__ inputs,
                   const float* __restrict__ W1, const float* __restrict__ b1,
                   const float* __restrict__ tau1,
                   const float* __restrict__ W2, const float* __restrict__ b2,
                   const float* __restrict__ tau2,
                   const float* __restrict__ W3, const float* __restrict__ b3,
                   const float* __restrict__ tau3,
                   float* __restrict__ out)
{
    // ---------------- LDS (weights + full input tile resident) ----------------
    __shared__ __align__(16) _Float16 W1f[H1N][H1N];     // recurrent part [n][k], 8 KB
    __shared__ __align__(16) _Float16 W2f[H2N][KTOT];    // [n][k], 160 KB
    __shared__ __align__(16) _Float16 W3f[16][H2N];      // padded to 16 rows, 8 KB
    __shared__ __align__(16) _Float16 s12[TILE_B][KTOT]; // spikes [s1|s2], 10 KB
    __shared__ __align__(16) float    xT[TSTEPS][TILE_B];// whole input tile, 64 KB
    __shared__ float w1c0[H1N];                          // W1 input column
    __shared__ float a1L[H1N], b1L[H1N];
    __shared__ float a2L[H2N], b2L[H2N];
    __shared__ float a3L[16],  b3L[16];

    const int tid  = threadIdx.x;
    const int lane = tid & 31;
    const int wave = tid >> 5;
    const int b0   = blockIdx.x * TILE_B;       // batch-row base of this WG

    // ---------------- one-time init: weight staging (f32 -> f16) ----------------
    for (int i = tid; i < H1N * H1N; i += NTHREADS) {
        int n = i >> 6, k = i & 63;
        W1f[n][k] = (_Float16)W1[n * (H1N + 1) + 1 + k];   // skip input col 0
    }
    for (int i = tid; i < H2N * KTOT; i += NTHREADS)
        ((_Float16*)W2f)[i] = (_Float16)W2[i];             // same [n][k] layout
    for (int i = tid; i < 16 * H2N; i += NTHREADS) {
        int n = i >> 8, k = i & 255;
        ((_Float16*)W3f)[i] = (n < OUTN) ? (_Float16)W3[n * H2N + k]
                                         : (_Float16)0.0f;
    }
    // whole 16x1024 input tile -> LDS (coalesced over t), transposed to [t][row]
    for (int i = tid; i < TILE_B * TSTEPS; i += NTHREADS) {
        int row = i >> 10, t = i & (TSTEPS - 1);
        xT[t][row] = inputs[(b0 + row) * TSTEPS + t];      // [B,T,1]
    }
    for (int n = tid; n < H1N; n += NTHREADS) {
        w1c0[n] = W1[n * (H1N + 1)];
        a1L[n]  = __expf(-1.0f / tau1[n]);
        b1L[n]  = b1[n];
    }
    for (int n = tid; n < H2N; n += NTHREADS) {
        a2L[n] = __expf(-1.0f / tau2[n]);
        b2L[n] = b2[n];
    }
    if (tid < 16) {
        a3L[tid] = (tid < OUTN) ? __expf(-1.0f / tau3[tid]) : 0.0f;
        b3L[tid] = (tid < OUTN) ? b3[tid] : 0.0f;
    }
    for (int i = tid; i < TILE_B * KTOT; i += NTHREADS)
        ((_Float16*)s12)[i] = (_Float16)0.0f;              // s1=s2=0 at t=0
    __syncthreads();

    // ---------------- per-lane constants (C-layout: col = lane&15) ----------------
    const int  col  = lane & 15;
    const int  hi8  = (lane & 16) ? 8 : 0;   // row offset of this half-wave
    const int  nb1  = TILE_B * wave;         // this wave's layer-1 n-tile
    const float a1v = a1L[nb1 + col];
    const float b1v = b1L[nb1 + col];
    const float w1v = w1c0[nb1 + col];
    float a2v[4], b2v[4];
#pragma unroll
    for (int jj = 0; jj < 4; ++jj) {
        int n = (4 * wave + jj) * 16 + col;  // this wave's layer-2 neurons
        a2v[jj] = a2L[n];
        b2v[jj] = b2L[n];
    }
    const float a3v = a3L[col];
    const float b3v = b3L[col];
    const bool  outMask = (col < OUTN);

    // ---------------- recurrent state (WMMA C layout, f32) ----------------
    v8f m1 = {}, sp1 = {};
    v8f m2[4] = {}, sp2[4] = {};
    v8f m3 = {}, sp3 = {}, accO = {};        // wave 0 only

    // ---------------- time loop (serial dependency; no VMEM inside) ----------------
    for (int t = 0; t < TSTEPS; ++t) {
        // ---- layer 1: d1 = s1_prev @ W1rec^T  (16x64x64) ----
        v8f d1 = {};
#pragma unroll
        for (int kc = 0; kc < H1N / 32; ++kc) {
            v16h a = ld_frag(&s12[0][0], KTOT, 0, kc * 32, lane);
            v16h b = ld_frag(&W1f[0][0], H1N, nb1, kc * 32, lane);
            d1 = wmma16(a, b, d1);
        }
        // x rows needed by this half-wave are contiguous in LDS: 2x 16B loads
        v4f xlo = *(const v4f*)&xT[t][hi8];
        v4f xhi = *(const v4f*)&xT[t][hi8 + 4];
        v8f ns1 = {};
#pragma unroll
        for (int r = 0; r < 8; ++r) {
            float xv = (r < 4) ? xlo[r & 3] : xhi[r & 3];
            float d  = d1[r] + xv * w1v + b1v;             // + x*W1[:,0] + bias
            float m  = m1[r] * a1v + (1.0f - a1v) * d - sp1[r];  // vth_rnn*dt = 1
            m1[r]  = m;
            ns1[r] = (m - 1.0f > 0.0f) ? 1.0f : 0.0f;
        }
        sp1 = ns1;
        __syncthreads();                                   // B1: s1 reads done
#pragma unroll
        for (int r = 0; r < 8; ++r)
            s12[hi8 + r][nb1 + col] = (_Float16)ns1[r];
        __syncthreads();                                   // B2: s1 published

        // ---- layer 2: d2 = [s1_new, s2_prev] @ W2^T  (16x320x256) ----
        v8f d2[4] = {};
        for (int kc = 0; kc < KTOT / 32; ++kc) {
            v16h a = ld_frag(&s12[0][0], KTOT, 0, kc * 32, lane);
#pragma unroll
            for (int jj = 0; jj < 4; ++jj) {
                v16h b = ld_frag(&W2f[0][0], KTOT, (4 * wave + jj) * 16,
                                 kc * 32, lane);
                d2[jj] = wmma16(a, b, d2[jj]);
            }
        }
#pragma unroll
        for (int jj = 0; jj < 4; ++jj) {
#pragma unroll
            for (int r = 0; r < 8; ++r) {
                float d = d2[jj][r] + b2v[jj];
                float m = m2[jj][r] * a2v[jj] + (1.0f - a2v[jj]) * d - sp2[jj][r];
                m2[jj][r]  = m;
                sp2[jj][r] = (m - 1.0f > 0.0f) ? 1.0f : 0.0f;
            }
        }
        __syncthreads();                                   // B3: s2 reads done
#pragma unroll
        for (int jj = 0; jj < 4; ++jj)
#pragma unroll
            for (int r = 0; r < 8; ++r)
                s12[hi8 + r][H1N + (4 * wave + jj) * 16 + col] =
                    (_Float16)sp2[jj][r];
        __syncthreads();                                   // B4: s2 published

        // ---- layer 3 (wave 0): d3 = s2_new @ W3^T (16x256x16, cols<10 live) ----
        if (wave == 0) {
            v8f d3 = {};
#pragma unroll
            for (int kc = 0; kc < H2N / 32; ++kc) {
                v16h a = ld_frag(&s12[0][0], KTOT, 0, H1N + kc * 32, lane);
                v16h b = ld_frag(&W3f[0][0], H2N, 0, kc * 32, lane);
                d3 = wmma16(a, b, d3);
            }
#pragma unroll
            for (int r = 0; r < 8; ++r) {
                float d = d3[r] + b3v;
                float m = m3[r] * a3v + (1.0f - a3v) * d - 0.5f * sp3[r];
                m3[r]   = m;
                float s = (outMask && (m - 0.5f > 0.0f)) ? 1.0f : 0.0f;
                sp3[r]  = s;
                accO[r] += s;
            }
        }
        // next-iteration B1 orders layer-3 reads vs. future s12 writes
    }

    // ---------------- epilogue: summed output spikes ----------------
    if (wave == 0 && outMask) {
#pragma unroll
        for (int r = 0; r < 8; ++r)
            out[(b0 + hi8 + r) * OUTN + col] = accO[r];
    }
}

extern "C" void kernel_launch(void* const* d_in, const int* in_sizes, int n_in,
                              void* d_out, int out_size, void* d_ws, size_t ws_size,
                              hipStream_t stream) {
    (void)in_sizes; (void)n_in; (void)out_size; (void)d_ws; (void)ws_size;
    // setup_inputs order: inputs, labels, W1, b1, tau1, W2, b2, tau2, W3, b3, tau3
    const float* inputs = (const float*)d_in[0];
    const float* W1   = (const float*)d_in[2];
    const float* b1   = (const float*)d_in[3];
    const float* tau1 = (const float*)d_in[4];
    const float* W2   = (const float*)d_in[5];
    const float* b2   = (const float*)d_in[6];
    const float* tau2 = (const float*)d_in[7];
    const float* W3   = (const float*)d_in[8];
    const float* b3   = (const float*)d_in[9];
    const float* tau3 = (const float*)d_in[10];
    float* out = (float*)d_out;

    dim3 grid(BATCH / TILE_B);   // 16 independent batch tiles
    dim3 block(NTHREADS);        // 4 waves of 32 (wave32)
    hipLaunchKernelGGL(snn_forward_kernel, grid, block, 0, stream,
                       inputs, W1, b1, tau1, W2, b2, tau2, W3, b3, tau3, out);
}
